// LoopedMambaLayer_28802050687824
// MI455X (gfx1250) — matrix operand order repaired and checked by reading, not compiled, not run on hardware
//
#include <hip/hip_runtime.h>

#define DMODEL 1024
#define DINNER 2048
#define DCONV 4
#define DSTATE 16
#define DTRANK 64
#define CHUNKC 64
#define ALPHA_C 0.1f
#define BB 2
#define TT 2048
#define BT (BB * TT)

typedef __attribute__((ext_vector_type(16))) __bf16 v16bf;
typedef __attribute__((ext_vector_type(8)))  float v8f;

__device__ __forceinline__ float4 ldg4(const float* p) { return *(const float4*)p; }

// ---------------- WMMA bf16 fragment loaders (contiguous k = 16*(lane/16) + s, same order in A and B) ----

// A 16x32 (MxK): lane<16 -> halves 0..15 = k0..k0+15
//                lane>=16 -> same + 16
template <int TA, int KS>
__device__ __forceinline__ v16bf load_a_frag(const float* __restrict__ A, long lda,
                                             int m0, int k0, int lane,
                                             const float* __restrict__ kScale) {
  int mr = m0 + (lane & 15);
  int kb = (lane & 16) ? 16 : 0;
  v16bf a;
  if constexpr (!TA && !KS) {
    const float* p0 = A + (long)mr * lda + k0 + kb;
    float4 x0 = ldg4(p0), x1 = ldg4(p0 + 4), x2 = ldg4(p0 + 8), x3 = ldg4(p0 + 12);
    a[0] = (__bf16)x0.x;  a[1] = (__bf16)x0.y;  a[2] = (__bf16)x0.z;  a[3] = (__bf16)x0.w;
    a[4] = (__bf16)x1.x;  a[5] = (__bf16)x1.y;  a[6] = (__bf16)x1.z;  a[7] = (__bf16)x1.w;
    a[8] = (__bf16)x2.x;  a[9] = (__bf16)x2.y;  a[10] = (__bf16)x2.z; a[11] = (__bf16)x2.w;
    a[12] = (__bf16)x3.x; a[13] = (__bf16)x3.y; a[14] = (__bf16)x3.z; a[15] = (__bf16)x3.w;
  } else {
#pragma unroll
    for (int h = 0; h < 16; ++h) {
      int k = k0 + kb + h;
      float v = TA ? A[(long)k * lda + mr] : A[(long)mr * lda + k];
      if constexpr (KS) v *= kScale[k];
      a[h] = (__bf16)v;
    }
  }
  return a;
}

// B 32x16 (KxN): n = lane&15; half h -> k = k0 + h + (lane>=16 ? 16 : 0)
template <int TB>
__device__ __forceinline__ v16bf load_b_frag(const float* __restrict__ Bm, long ldb,
                                             int k0, int n0, int lane) {
  int n = n0 + (lane & 15);
  int kb = (lane & 16) ? 16 : 0;
  v16bf b;
  if constexpr (TB) {  // B given as NxK row-major (weight W, y = x @ W^T): 16 contiguous
    const float* p0 = Bm + (long)n * ldb + k0 + kb;
    float4 x0 = ldg4(p0), x1 = ldg4(p0 + 4), x2 = ldg4(p0 + 8), x3 = ldg4(p0 + 12);
    b[0] = (__bf16)x0.x;  b[1] = (__bf16)x0.y;  b[2] = (__bf16)x0.z;  b[3] = (__bf16)x0.w;
    b[4] = (__bf16)x1.x;  b[5] = (__bf16)x1.y;  b[6] = (__bf16)x1.z;  b[7] = (__bf16)x1.w;
    b[8] = (__bf16)x2.x;  b[9] = (__bf16)x2.y;  b[10] = (__bf16)x2.z; b[11] = (__bf16)x2.w;
    b[12] = (__bf16)x3.x; b[13] = (__bf16)x3.y; b[14] = (__bf16)x3.z; b[15] = (__bf16)x3.w;
  } else {             // B given as KxN row-major: strided
#pragma unroll
    for (int h = 0; h < 16; ++h)
      b[h] = (__bf16)Bm[(long)(k0 + kb + h) * ldb + n];
  }
  return b;
}

// --- Generic WMMA GEMM: C = alpha*rowScale*elemScale*(A@B) [+bias] [+beta*Cin], act
// Block = 4 waves = 64(M) x 64(N) tile. B fragments staged once per block into LDS
// (bf16, fragment layout) and shared by all 4 waves: 4x cut in global B traffic.
template <int TA, int TB, int KS>
__global__ void wmma_gemm_t(const float* __restrict__ A, long lda, long sAb,
                            const float* __restrict__ aKScale,
                            const float* __restrict__ Bm, long ldb, long sBb,
                            const float* __restrict__ bias,
                            const float* __restrict__ Cin, long ldcin, long sCin,
                            float beta, const float* __restrict__ betaPtr,
                            const float* __restrict__ elemScale,   // [M*N] or null
                            const float* __restrict__ rowScale,    // [M] or null
                            float alpha, int act,                  // 0 none, 1 softplus
                            float* __restrict__ C, long ldc, long sCb,
                            int M, int N, int K) {
  __shared__ v16bf ldsB[4][32];                 // 4 KB: four 32x16 bf16 B fragments

  int lane = threadIdx.x & 31;
  int wave = threadIdx.x >> 5;
  int m0 = blockIdx.y * 64 + wave * 16;         // M always a multiple of 64 here
  int nb = blockIdx.x * 64;
  long bz = blockIdx.z;
  const float* Ab = A + bz * sAb;
  const float* Bb = Bm + bz * sBb;
  const float* Cb = Cin ? (Cin + bz * sCin) : nullptr;
  float* Co = C + bz * sCb;

  int nt[4];
#pragma unroll
  for (int j = 0; j < 4; ++j) {
    int n0 = nb + j * 16;
    nt[j] = (n0 > N - 16) ? (N - 16) : n0;      // clamp (only pure-write GEMMs hit this)
  }

  v8f acc[4];
#pragma unroll
  for (int j = 0; j < 4; ++j) acc[j] = (v8f){0.f,0.f,0.f,0.f,0.f,0.f,0.f,0.f};

  for (int k0 = 0; k0 < K; k0 += 32) {
    // wave j stages B tile j (converted to bf16, in WMMA fragment layout)
    ldsB[wave][lane] = load_b_frag<TB>(Bb, ldb, k0, nt[wave], lane);
    __syncthreads();
    v16bf af = load_a_frag<TA, KS>(Ab, lda, m0, k0, lane, aKScale);
#pragma unroll
    for (int j = 0; j < 4; ++j) {
      v16bf bf = ldsB[j][lane];
      acc[j] = __builtin_amdgcn_wmma_f32_16x16x32_bf16(false, af, false, bf,
                                                       (short)0, acc[j], false, false);
    }
    __syncthreads();
  }

  float be = beta;
  if (betaPtr) be *= betaPtr[0];
#pragma unroll
  for (int j = 0; j < 4; ++j) {
    int n = nt[j] + (lane & 15);
#pragma unroll
    for (int r = 0; r < 8; ++r) {
      int m = m0 + r + ((lane & 16) ? 8 : 0);
      float v = acc[j][r] * alpha;
      if (rowScale)  v *= rowScale[m];
      if (elemScale) v *= elemScale[(long)m * N + n];
      if (bias)      v += bias[n];
      if (Cb)        v += be * Cb[(long)m * ldcin + n];
      if (act == 1)  v = (v > 20.f) ? v : log1pf(__expf(v));
      Co[(long)m * ldc + n] = v;
    }
  }
}

// ---------------- elementwise / recurrent kernels (f32 VALU) ---------------------

__global__ void rmsnorm_kernel(const float* __restrict__ x, const float* __restrict__ w,
                               float* __restrict__ o, int D) {
  long row = blockIdx.x;
  const float* xr = x + row * (long)D;
  float ss = 0.f;
  for (int i = threadIdx.x; i < D; i += blockDim.x) { float v = xr[i]; ss += v * v; }
  __shared__ float sh[256];
  sh[threadIdx.x] = ss;
  __syncthreads();
  for (int st = 128; st > 0; st >>= 1) {
    if ((int)threadIdx.x < st) sh[threadIdx.x] += sh[threadIdx.x + st];
    __syncthreads();
  }
  float scale = rsqrtf(sh[0] / (float)D + 1e-5f);
  for (int i = threadIdx.x; i < D; i += blockDim.x)
    o[row * (long)D + i] = xr[i] * scale * w[i];
}

__global__ void conv_silu_kernel(const float* __restrict__ xz, const float* __restrict__ cw,
                                 const float* __restrict__ cb, float* __restrict__ xma) {
  long i = (long)blockIdx.x * blockDim.x + threadIdx.x;
  long total = (long)BT * DINNER;
  if (i >= total) return;
  int d = (int)(i % DINNER);
  long bt = i / DINNER;
  int t = (int)(bt % TT);
  int b = (int)(bt / TT);
  float acc = cb[d];
#pragma unroll
  for (int j = 0; j < DCONV; ++j) {
    int tt = t - (DCONV - 1) + j;
    if (tt >= 0)
      acc += xz[((long)(b * TT + tt)) * (2 * DINNER) + d] * cw[d * DCONV + j];
  }
  xma[i] = acc / (1.f + __expf(-acc));
}

__global__ void scan_kernel(const float* __restrict__ delta, const float* __restrict__ xma,
                            const float* __restrict__ xdbl, const float* __restrict__ xz,
                            const float* __restrict__ A_log, const float* __restrict__ D_ssm,
                            float* __restrict__ y) {
  int idx = blockIdx.x * blockDim.x + threadIdx.x;
  if (idx >= BB * DINNER) return;
  int b = idx / DINNER, d = idx % DINNER;
  float Av[DSTATE], h[DSTATE];
#pragma unroll
  for (int s = 0; s < DSTATE; ++s) { Av[s] = -__expf(A_log[d * DSTATE + s]); h[s] = 0.f; }
  float Dd = D_ssm[d];
  for (int t = 0; t < TT; ++t) {
    long row = (long)b * TT + t;
    float dt = delta[row * DINNER + d];
    float u  = xma[row * DINNER + d];
    const float* bt = xdbl + row * 96 + DTRANK;
    const float* ct = xdbl + row * 96 + DTRANK + DSTATE;
    float du = dt * u;
    float yv = 0.f;
#pragma unroll
    for (int s = 0; s < DSTATE; ++s) {
      h[s] = h[s] * __expf(dt * Av[s]) + du * bt[s];
      yv += h[s] * ct[s];
    }
    yv += u * Dd;
    float r = xz[row * (2 * DINNER) + DINNER + d];
    yv *= r / (1.f + __expf(-r));
    y[row * DINNER + d] = yv;
  }
}

__global__ void vec_add(const float* __restrict__ a, const float* __restrict__ b,
                        float* __restrict__ o, long n) {
  long i = (long)blockIdx.x * blockDim.x + threadIdx.x;
  if (i < n) o[i] = a[i] + b[i];
}

__global__ void shift_kernel(const float* __restrict__ src, float* __restrict__ dst) {
  long i = (long)blockIdx.x * blockDim.x + threadIdx.x;
  long total = (long)BT * DMODEL;
  if (i >= total) return;
  long bt = i / DMODEL;
  int t = (int)(bt % TT);
  dst[i] = (t == 0) ? 0.f : src[i - DMODEL];
}

__global__ void fill_zero(float* __restrict__ p, long n) {
  long i = (long)blockIdx.x * blockDim.x + threadIdx.x;
  if (i < n) p[i] = 0.f;
}

__global__ void attend_init(const float* __restrict__ decay, float* __restrict__ gin,
                            float* __restrict__ gw, float* __restrict__ gc,
                            float* __restrict__ Mm) {
  float gamma = 1.f / (1.f + __expf(-decay[0]));
  int c = threadIdx.x;
  if (c < CHUNKC) {
    gin[c] = powf(gamma, (float)c);
    gw[c]  = powf(gamma, (float)(CHUNKC - 1 - c));
    for (int k = 0; k < CHUNKC; ++k)
      Mm[c * CHUNKC + k] = (c > k) ? powf(gamma, (float)(c - 1 - k)) : 0.f;
  }
  if (c == 0) gc[0] = powf(gamma, (float)CHUNKC);
}

// ---------------- host-side launch helpers -------------------------------------

static inline void gemm_launch(hipStream_t s,
    const float* A, long lda, long sAb, int tA, const float* aK,
    const float* Bm, long ldb, long sBb, int tB,
    const float* bias,
    const float* Cin, long ldcin, long sCin, float beta, const float* betaPtr,
    const float* elemScale, const float* rowScale,
    float alpha, int act,
    float* C, long ldc, long sCb, int M, int N, int K, int batch) {
  dim3 grid((N + 63) / 64, (M + 63) / 64, batch);
  if (!tA && tB)
    wmma_gemm_t<0, 1, 0><<<grid, 128, 0, s>>>(A, lda, sAb, aK, Bm, ldb, sBb, bias,
        Cin, ldcin, sCin, beta, betaPtr, elemScale, rowScale, alpha, act,
        C, ldc, sCb, M, N, K);
  else if (!tA && !tB)
    wmma_gemm_t<0, 0, 0><<<grid, 128, 0, s>>>(A, lda, sAb, aK, Bm, ldb, sBb, bias,
        Cin, ldcin, sCin, beta, betaPtr, elemScale, rowScale, alpha, act,
        C, ldc, sCb, M, N, K);
  else
    wmma_gemm_t<1, 0, 1><<<grid, 128, 0, s>>>(A, lda, sAb, aK, Bm, ldb, sBb, bias,
        Cin, ldcin, sCin, beta, betaPtr, elemScale, rowScale, alpha, act,
        C, ldc, sCb, M, N, K);
}

static void run_mamba(hipStream_t st, const float* xin, const float* normw,
                      const float* inW, const float* inB, const float* convW,
                      const float* convB, const float* xpW, const float* dtW,
                      const float* dtB, const float* Alog, const float* Dssm,
                      const float* outW, const float* outB,
                      float* xn, float* xz, float* xma, float* xdbl,
                      float* delta, float* ybuf, float* mout) {
  rmsnorm_kernel<<<BT, 256, 0, st>>>(xin, normw, xn, DMODEL);
  // xz = xn @ inW^T + inB   (4096 x 4096 x 1024)
  gemm_launch(st, xn, DMODEL, 0, 0, nullptr, inW, DMODEL, 0, 1, inB,
              nullptr, 0, 0, 0.f, nullptr, nullptr, nullptr, 1.f, 0,
              xz, 2 * DINNER, 0, BT, 2 * DINNER, DMODEL, 1);
  long tot = (long)BT * DINNER;
  conv_silu_kernel<<<(unsigned)((tot + 255) / 256), 256, 0, st>>>(xz, convW, convB, xma);
  // xdbl = xma @ xpW^T      (4096 x 96 x 2048)
  gemm_launch(st, xma, DINNER, 0, 0, nullptr, xpW, DINNER, 0, 1, nullptr,
              nullptr, 0, 0, 0.f, nullptr, nullptr, nullptr, 1.f, 0,
              xdbl, 96, 0, BT, 96, DINNER, 1);
  // delta = softplus(xdbl[:, :64] @ dtW^T + dtB)  (4096 x 2048 x 64)
  gemm_launch(st, xdbl, 96, 0, 0, nullptr, dtW, DTRANK, 0, 1, dtB,
              nullptr, 0, 0, 0.f, nullptr, nullptr, nullptr, 1.f, 1,
              delta, DINNER, 0, BT, DINNER, DTRANK, 1);
  scan_kernel<<<(BB * DINNER + 255) / 256, 256, 0, st>>>(delta, xma, xdbl, xz,
                                                         Alog, Dssm, ybuf);
  // mout = ybuf @ outW^T + outB  (4096 x 1024 x 2048)
  gemm_launch(st, ybuf, DINNER, 0, 0, nullptr, outW, DINNER, 0, 1, outB,
              nullptr, 0, 0, 0.f, nullptr, nullptr, nullptr, 1.f, 0,
              mout, DMODEL, 0, BT, DMODEL, DINNER, 1);
}

extern "C" void kernel_launch(void* const* d_in, const int* in_sizes, int n_in,
                              void* d_out, int out_size, void* d_ws, size_t ws_size,
                              hipStream_t stream) {
  const float* x      = (const float*)d_in[0];
  const float* n1w    = (const float*)d_in[1];
  const float* n2w    = (const float*)d_in[2];
  const float* inW    = (const float*)d_in[3];
  const float* inB    = (const float*)d_in[4];
  const float* convW  = (const float*)d_in[5];
  const float* convB  = (const float*)d_in[6];
  const float* xpW    = (const float*)d_in[7];
  const float* dtW    = (const float*)d_in[8];
  const float* dtB    = (const float*)d_in[9];
  const float* Alog   = (const float*)d_in[10];
  const float* Dssm   = (const float*)d_in[11];
  const float* outW   = (const float*)d_in[12];
  const float* outB   = (const float*)d_in[13];
  const float* wWrite = (const float*)d_in[14];
  const float* wRead  = (const float*)d_in[15];
  const float* decay  = (const float*)d_in[16];
  float* out = (float*)d_out;
  (void)in_sizes; (void)n_in; (void)out_size;

  float* p    = (float*)d_ws;
  float* xn   = p; p += (long)BT * DMODEL;
  float* xz   = p; p += (long)BT * 2 * DINNER;
  float* xma  = p; p += (long)BT * DINNER;
  float* xdbl = p; p += (long)BT * 96;
  float* delta= p; p += (long)BT * DINNER;
  float* ybuf = p; p += (long)BT * DINNER;
  float* mout = p; p += (long)BT * DMODEL;
  float* x1   = p; p += (long)BT * DMODEL;
  float* Wst  = p; p += (long)BB * DMODEL * DMODEL;
  size_t need = (size_t)(p - (float*)d_ws) * sizeof(float);
  if (ws_size < need) return;

  // attend-phase buffer reuse (producers dead by then)
  float* vbuf  = ybuf;
  float* wkb   = delta;
  float* reads = xma;
  float* SM    = xz;
  float* gin   = xz + 8192;
  float* gwv   = gin + CHUNKC;
  float* gamC  = gwv + CHUNKC;
  float* Mmask = gamC + 1;
  float* attn  = xn;

  const long nOut = (long)BT * DMODEL;
  const long sB   = (long)TT * DMODEL;
  const long sW   = (long)DMODEL * DMODEL;

  // ---- block 1 ----
  run_mamba(stream, x, n1w, inW, inB, convW, convB, xpW, dtW, dtB, Alog, Dssm,
            outW, outB, xn, xz, xma, xdbl, delta, ybuf, mout);
  vec_add<<<(unsigned)((nOut + 255) / 256), 256, 0, stream>>>(x, mout, x1, nOut);

  // ---- block 2 ----
  run_mamba(stream, x1, n2w, inW, inB, convW, convB, xpW, dtW, dtB, Alog, Dssm,
            outW, outB, xn, xz, xma, xdbl, delta, ybuf, mout);   // mout = out2

  // ---- memory_attend ----
  gemm_launch(stream, mout, DMODEL, 0, 0, nullptr, wWrite, DMODEL, 0, 1, nullptr,
              nullptr, 0, 0, 0.f, nullptr, nullptr, nullptr, 1.f, 0,
              vbuf, DMODEL, 0, BT, DMODEL, DMODEL, 1);
  shift_kernel<<<(unsigned)((nOut + 255) / 256), 256, 0, stream>>>(mout, wkb);
  fill_zero<<<(unsigned)((BB * sW + 255) / 256), 256, 0, stream>>>(Wst, BB * sW);
  attend_init<<<1, 64, 0, stream>>>(decay, gin, gwv, gamC, Mmask);

  for (int ch = 0; ch < TT / CHUNKC; ++ch) {
    const float* rk = mout + (long)ch * CHUNKC * DMODEL;
    const float* wc = wkb  + (long)ch * CHUNKC * DMODEL;
    const float* vc = vbuf + (long)ch * CHUNKC * DMODEL;
    float*       rc = reads + (long)ch * CHUNKC * DMODEL;
    // SM = (rk @ wk^T) ∘ M   (64 x 64 x 1024, batch 2)
    gemm_launch(stream, rk, DMODEL, sB, 0, nullptr, wc, DMODEL, sB, 1, nullptr,
                nullptr, 0, 0, 0.f, nullptr, Mmask, nullptr, 1.f, 0,
                SM, CHUNKC, (long)CHUNKC * CHUNKC, CHUNKC, CHUNKC, DMODEL, BB);
    // intra: reads_c = SM @ v_c   (64 x 1024 x 64)
    gemm_launch(stream, SM, CHUNKC, (long)CHUNKC * CHUNKC, 0, nullptr,
                vc, DMODEL, sB, 0, nullptr,
                nullptr, 0, 0, 0.f, nullptr, nullptr, nullptr, 1.f, 0,
                rc, DMODEL, sB, CHUNKC, DMODEL, CHUNKC, BB);
    // inter: reads_c += g_in[c] * (rk @ W^T)   (64 x 1024 x 1024)
    gemm_launch(stream, rk, DMODEL, sB, 0, nullptr, Wst, DMODEL, sW, 1, nullptr,
                rc, DMODEL, sB, 1.f, nullptr, nullptr, gin, 1.f, 0,
                rc, DMODEL, sB, CHUNKC, DMODEL, DMODEL, BB);
    // W = gamma^C * W + (v_c * g_w)^T @ wk_c   (1024 x 1024 x 64, in place)
    gemm_launch(stream, vc, DMODEL, sB, 1, gwv, wc, DMODEL, sB, 0, nullptr,
                Wst, DMODEL, sW, 1.f, gamC, nullptr, nullptr, 1.f, 0,
                Wst, DMODEL, sW, DMODEL, DMODEL, CHUNKC, BB);
  }

  // attn = out2 + ALPHA * (reads @ wRead^T)
  gemm_launch(stream, reads, DMODEL, 0, 0, nullptr, wRead, DMODEL, 0, 1, nullptr,
              mout, DMODEL, 0, 1.f, nullptr, nullptr, nullptr, ALPHA_C, 0,
              attn, DMODEL, 0, BT, DMODEL, DMODEL, 1);
  vec_add<<<(unsigned)((nOut + 255) / 256), 256, 0, stream>>>(x1, attn, out, nOut);
}
